// TransitionUpConv_1881195675672
// MI455X (gfx1250) — matrix operand order repaired and checked by reading, not compiled, not run on hardware
//
#include <hip/hip_runtime.h>
#include <hip/hip_bf16.h>

typedef __attribute__((ext_vector_type(16))) _Float16 v16h;
typedef __attribute__((ext_vector_type(8)))  float    v8f;

#define NBATCH 2
#define CIN1   64
#define CH     32
#define COUT   32
#define DDIM   40
#define SVOX   64000      // 40*40*40
#define NC     100000
#define NF     400000
#define KST    7
#define GEPS   1e-5f

// ---------------------------------------------------------------------------
// K1: y_raw[b][o][s] = sum_c w_reduce[o][c] * x1[b][c][s]   (32x64 GEMM, VALU)
// ---------------------------------------------------------------------------
__global__ void reduce_gemm(const float* __restrict__ x1,
                            const float* __restrict__ wr,
                            float* __restrict__ y_raw) {
    __shared__ float wl[CH * CIN1];
    for (int i = threadIdx.x; i < CH * CIN1; i += blockDim.x) wl[i] = wr[i];
    __syncthreads();
    int gid = blockIdx.x * blockDim.x + threadIdx.x;   // over B*S
    if (gid >= NBATCH * SVOX) return;
    int b = gid / SVOX, s = gid - b * SVOX;
    float acc[CH];
#pragma unroll
    for (int o = 0; o < CH; ++o) acc[o] = 0.f;
    for (int c = 0; c < CIN1; ++c) {
        float xv = x1[((size_t)b * CIN1 + c) * SVOX + s];
#pragma unroll
        for (int o = 0; o < CH; ++o) acc[o] += wl[o * CIN1 + c] * xv;
    }
    for (int o = 0; o < CH; ++o)
        y_raw[((size_t)b * CH + o) * SVOX + s] = acc[o];
}

// ---------------------------------------------------------------------------
// Deterministic two-stage per-channel stats (sum, sumsq) for GroupNorm
// grid = (G, NCHUNK); part[(g*NCHUNK+chunk)*2 + {0,1}]
// ---------------------------------------------------------------------------
__global__ void stats_partial(const float* __restrict__ x, int L,
                              float* __restrict__ part) {
    int g = blockIdx.x, chunk = blockIdx.y;
    const float* p = x + (size_t)g * L;
    float s = 0.f, ss = 0.f;
    for (int i = chunk * blockDim.x + threadIdx.x; i < L;
         i += gridDim.y * blockDim.x) {
        float v = p[i];
        s += v; ss += v * v;
    }
    __shared__ float r0[256], r1[256];
    r0[threadIdx.x] = s; r1[threadIdx.x] = ss;
    __syncthreads();
    for (int w = 128; w > 0; w >>= 1) {
        if ((int)threadIdx.x < w) {
            r0[threadIdx.x] += r0[threadIdx.x + w];
            r1[threadIdx.x] += r1[threadIdx.x + w];
        }
        __syncthreads();
    }
    if (threadIdx.x == 0) {
        part[((size_t)g * gridDim.y + chunk) * 2 + 0] = r0[0];
        part[((size_t)g * gridDim.y + chunk) * 2 + 1] = r1[0];
    }
}

__global__ void stats_final(const float* __restrict__ part, int nchunk,
                            float* __restrict__ stats, int G) {
    int i = blockIdx.x * blockDim.x + threadIdx.x;
    if (i >= G * 2) return;
    int g = i >> 1, which = i & 1;
    float s = 0.f;
    for (int c = 0; c < nchunk; ++c) s += part[((size_t)g * nchunk + c) * 2 + which];
    stats[g * 2 + which] = s;
}

// ---------------------------------------------------------------------------
// K3: per fine particle n: GN+ReLU of y_raw, gathered through
// sample_idx[upsample_idx[n]], concat with x2, f32->f16, layout [b][n][64]
// ---------------------------------------------------------------------------
__global__ void pack_xcat(const float* __restrict__ x2,
                          const float* __restrict__ y_raw,
                          const float* __restrict__ stats_y,
                          const float* __restrict__ gamma,
                          const float* __restrict__ beta,
                          const int* __restrict__ samp,
                          const int* __restrict__ up,
                          _Float16* __restrict__ xcat) {
    int gid = blockIdx.x * blockDim.x + threadIdx.x;   // over B*NF
    if (gid >= NBATCH * NF) return;
    int b = gid / NF, n = gid - b * NF;
    int j = up[(size_t)b * NF + n];
    int s = samp[(size_t)b * NC + j];
    _Float16* dst = xcat + (size_t)gid * (2 * CH);
#pragma unroll 4
    for (int c = 0; c < CH; ++c)
        dst[c] = (_Float16)x2[((size_t)b * CH + c) * NF + n];
#pragma unroll 4
    for (int c = 0; c < CH; ++c) {
        int g = b * CH + c;
        float mean = stats_y[g * 2] * (1.f / SVOX);
        float var  = stats_y[g * 2 + 1] * (1.f / SVOX) - mean * mean;
        float v = (y_raw[(size_t)g * SVOX + s] - mean) * gamma[c] *
                      rsqrtf(var + GEPS) + beta[c];
        dst[CH + c] = (_Float16)fmaxf(v, 0.f);
    }
}

// ---------------------------------------------------------------------------
// APR stencil conv as WMMA GEMM.
//   xin : [B][NF][CIN]  f16 (particle-major; one gather = CIN*2 contiguous B)
//   w   : [COUT][CIN][KST] f32
//   out : [B][COUT][NF] f32 (raw, pre-GroupNorm)
// Per wave: one 16-particle tile, both 16-row M-tiles of COUT=32.
// A fragments pre-staged in LDS as 16x32 tiles in this kernel's contiguous-K order:
//   lane<16 : M=lane,    halves 0-15 -> K 0-15
//   lane>=16: M=lane-16, halves 0-15 -> K 16-31
// B fragment (16-bit 32x16): lanes 0-15 col=lane K 0-15, lanes 16-31 K 16-31
//   -> each lane reads 32 contiguous bytes of one gathered particle.
// ---------------------------------------------------------------------------
template <int CIN>
__global__ void apr_conv_wmma(const _Float16* __restrict__ xin,
                              const float* __restrict__ w,
                              const int* __restrict__ nbr,
                              float* __restrict__ out_raw) {
    constexpr int KCH   = CIN / 32;            // K-chunks of 32
    constexpr int NFRAG = 2 * KST * KCH;       // (m, k, cc)
    __shared__ __align__(32) _Float16 aLDS[NFRAG][32][16];

    const int tid = threadIdx.x;
    // Stage f16 A fragments (weights) into LDS.
    for (int idx = tid; idx < NFRAG * 32 * 16; idx += blockDim.x) {
        int frag = idx >> 9;          // /512
        int rem  = idx & 511;
        int lane = rem >> 4;
        int jj   = rem & 15;
        int cc   = frag % KCH;
        int k    = (frag / KCH) % KST;
        int m    = frag / (KCH * KST);
        int M    = lane & 15;
        int kb   = (lane >> 4) * 8;
        int kidx = kb * 2 + jj;
        int c    = cc * 32 + kidx;
        int o    = m * 16 + M;
        aLDS[frag][lane][jj] = (_Float16)w[((size_t)o * CIN + c) * KST + k];
    }
    __syncthreads();

    const int lane = tid & 31;
    const int wave = tid >> 5;
    const int nsub = lane & 15;
    const int hsel = lane >> 4;                 // 0: K 0-15, 1: K 16-31
    const int tilesPerBatch = NF / 16;          // 25000
    const int totalTiles = NBATCH * tilesPerBatch;
    const int wavesPerGrid = gridDim.x * (blockDim.x >> 5);

    for (int t = blockIdx.x * (blockDim.x >> 5) + wave; t < totalTiles;
         t += wavesPerGrid) {
        int b  = t / tilesPerBatch;
        int nb = (t - b * tilesPerBatch) * 16;
        int n  = nb + nsub;
        v8f acc0 = {}; v8f acc1 = {};
        const size_t nbrBase = ((size_t)b * NF + n) * KST;
#pragma unroll
        for (int k = 0; k < KST; ++k) {
            int p = nbr[nbrBase + k];
            const _Float16* src = xin + ((size_t)b * NF + p) * CIN + hsel * 16;
#pragma unroll
            for (int cc = 0; cc < KCH; ++cc) {
                v16h bfrag = *(const v16h*)(src + cc * 32);
                v16h a0 = *(const v16h*)&aLDS[(0 * KST + k) * KCH + cc][lane][0];
                acc0 = __builtin_amdgcn_wmma_f32_16x16x32_f16(
                    false, a0, false, bfrag, (short)0, acc0, false, false);
                v16h a1 = *(const v16h*)&aLDS[(1 * KST + k) * KCH + cc][lane][0];
                acc1 = __builtin_amdgcn_wmma_f32_16x16x32_f16(
                    false, a1, false, bfrag, (short)0, acc1, false, false);
            }
        }
        // D layout: VGPR r, lanes 0-15: M=r, N=lane; lanes 16-31: M=r+8
        int ncol = nb + nsub;
#pragma unroll
        for (int r = 0; r < 8; ++r) {
            int o0 = r + hsel * 8;
            out_raw[((size_t)b * COUT + o0) * NF + ncol]        = acc0[r];
            out_raw[((size_t)b * COUT + 16 + o0) * NF + ncol]   = acc1[r];
        }
    }
}

// ---------------------------------------------------------------------------
// GN + ReLU of raw [b][c][n], emit f16 particle-major [b][n][COUT] for conv2
// ---------------------------------------------------------------------------
__global__ void norm_relu_to_f16(const float* __restrict__ raw,
                                 const float* __restrict__ stats,
                                 const float* __restrict__ gamma,
                                 const float* __restrict__ beta,
                                 _Float16* __restrict__ out) {
    int gid = blockIdx.x * blockDim.x + threadIdx.x;   // over B*NF
    if (gid >= NBATCH * NF) return;
    int b = gid / NF, n = gid - b * NF;
    _Float16* dst = out + (size_t)gid * COUT;
#pragma unroll 4
    for (int c = 0; c < COUT; ++c) {
        int g = b * COUT + c;
        float mean = stats[g * 2] * (1.f / NF);
        float var  = stats[g * 2 + 1] * (1.f / NF) - mean * mean;
        float v = (raw[(size_t)g * NF + n] - mean) * gamma[c] *
                      rsqrtf(var + GEPS) + beta[c];
        dst[c] = (_Float16)fmaxf(v, 0.f);
    }
}

// ---------------------------------------------------------------------------
// In-place GN + ReLU on d_out [b][c][n] f32
// ---------------------------------------------------------------------------
__global__ void norm_relu_inplace(float* __restrict__ data,
                                  const float* __restrict__ stats,
                                  const float* __restrict__ gamma,
                                  const float* __restrict__ beta) {
    size_t gid = (size_t)blockIdx.x * blockDim.x + threadIdx.x;
    if (gid >= (size_t)NBATCH * COUT * NF) return;
    size_t g = gid / NF;
    int c = (int)(g % COUT);
    float mean = stats[g * 2] * (1.f / NF);
    float var  = stats[g * 2 + 1] * (1.f / NF) - mean * mean;
    float v = (data[gid] - mean) * gamma[c] * rsqrtf(var + GEPS) + beta[c];
    data[gid] = fmaxf(v, 0.f);
}

// ---------------------------------------------------------------------------
extern "C" void kernel_launch(void* const* d_in, const int* in_sizes, int n_in,
                              void* d_out, int out_size, void* d_ws, size_t ws_size,
                              hipStream_t stream) {
    const float* x1       = (const float*)d_in[0];
    const float* x2       = (const float*)d_in[1];
    const float* w_reduce = (const float*)d_in[2];
    const float* gnr_g    = (const float*)d_in[3];
    const float* gnr_b    = (const float*)d_in[4];
    const float* w1       = (const float*)d_in[5];
    const float* g1       = (const float*)d_in[6];
    const float* b1       = (const float*)d_in[7];
    const float* w2       = (const float*)d_in[8];
    const float* g2       = (const float*)d_in[9];
    const float* b2       = (const float*)d_in[10];
    const int*   samp     = (const int*)d_in[11];
    const int*   up       = (const int*)d_in[12];
    const int*   nbr      = (const int*)d_in[13];
    float*       out      = (float*)d_out;

    auto au = [](size_t x) { return (x + 255) & ~(size_t)255; };
    char* wsb = (char*)d_ws;
    size_t off = 0;
    float*     y_raw  = (float*)(wsb + off);     off += au((size_t)NBATCH * CH * SVOX * 4);
    _Float16*  xcat   = (_Float16*)(wsb + off);  off += au((size_t)NBATCH * NF * 2 * CH * 2);
    float*     h1raw  = (float*)(wsb + off);     off += au((size_t)NBATCH * COUT * NF * 4);
    _Float16*  h1f16  = (_Float16*)(wsb + off);  off += au((size_t)NBATCH * NF * COUT * 2);
    float*     statsY = (float*)(wsb + off);     off += au(64 * 2 * 4);
    float*     stats1 = (float*)(wsb + off);     off += au(64 * 2 * 4);
    float*     stats2 = (float*)(wsb + off);     off += au(64 * 2 * 4);
    float*     part   = (float*)(wsb + off);     off += au(64 * 32 * 2 * 4);

    const int G = NBATCH * CH;        // 64 groups
    const int NCHUNK = 32;

    // 1) reduce_ch GEMM
    reduce_gemm<<<(NBATCH * SVOX + 255) / 256, 256, 0, stream>>>(x1, w_reduce, y_raw);

    // 2) GN stats over voxels
    stats_partial<<<dim3(G, NCHUNK), 256, 0, stream>>>(y_raw, SVOX, part);
    stats_final<<<1, 128, 0, stream>>>(part, NCHUNK, statsY, G);

    // 3) GN+ReLU + double gather + concat + f16 pack
    pack_xcat<<<(NBATCH * NF + 255) / 256, 256, 0, stream>>>(
        x2, y_raw, statsY, gnr_g, gnr_b, samp, up, xcat);

    // 4) APR conv 1 (Cin=64) via WMMA
    apr_conv_wmma<CIN1><<<(NBATCH * NF / 16 + 7) / 8, 256, 0, stream>>>(
        xcat, w1, nbr, h1raw);

    // 5) GN stats + normalize to f16 particle-major
    stats_partial<<<dim3(G, NCHUNK), 256, 0, stream>>>(h1raw, NF, part);
    stats_final<<<1, 128, 0, stream>>>(part, NCHUNK, stats1, G);
    norm_relu_to_f16<<<(NBATCH * NF + 255) / 256, 256, 0, stream>>>(
        h1raw, stats1, g1, b1, h1f16);

    // 6) APR conv 2 (Cin=32) via WMMA, raw straight into d_out
    apr_conv_wmma<COUT><<<(NBATCH * NF / 16 + 7) / 8, 256, 0, stream>>>(
        h1f16, w2, nbr, out);

    // 7) GN stats + in-place normalize/ReLU of final output
    stats_partial<<<dim3(G, NCHUNK), 256, 0, stream>>>(out, NF, part);
    stats_final<<<1, 128, 0, stream>>>(part, NCHUNK, stats2, G);
    norm_relu_inplace<<<((size_t)NBATCH * COUT * NF + 255) / 256, 256, 0, stream>>>(
        out, stats2, g2, b2);
}
